// ActionEncoder_49100066128401
// MI455X (gfx1250) — matrix operand order repaired and checked by reading, not compiled, not run on hardware
//
#include <hip/hip_runtime.h>
#include <cmath>

// CDNA5 / gfx1250: wave32, WMMA 16x16x32 f16->f32.
typedef __attribute__((ext_vector_type(16))) _Float16 v16h;
typedef __attribute__((ext_vector_type(8)))  _Float16 v8h;
typedef __attribute__((ext_vector_type(8)))  float    v8f;

#define WPB 8  // waves per block (256 threads)

// Branch-free tanh: v_med3 clamp + v_exp_f32 + v_rcp_f32 + fma; no divergence,
// no IEEE-division ladder. tanh saturates to +/-1 (f32) far before |x|=15.
__device__ __forceinline__ float fast_tanh(float x) {
    const float xc = fminf(fmaxf(x, -15.0f), 15.0f);
    const float e  = __expf(2.0f * xc);
    return 1.0f - 2.0f * __builtin_amdgcn_rcpf(e + 1.0f);
}

__global__ __launch_bounds__(256)
void action_encoder_wmma(const int*   __restrict__ idxs,   // [B,2] int32
                         const int*   __restrict__ types,  // [B]   int32
                         const float* __restrict__ W0,     // [4,64]
                         const float* __restrict__ b0,     // [4]
                         const float* __restrict__ W1,     // [4,128]
                         const float* __restrict__ b1,     // [4]
                         float*       __restrict__ out,    // [B,4]
                         int nB)
{
    // byte -> 8 x f16 one-hot expansion LUT (4 KB), + per-wave D staging tiles.
    __shared__ float4 lut[256];
    __shared__ float  stage[WPB][16 * 16];

    const int  lane   = threadIdx.x & 31;
    const int  wave   = threadIdx.x >> 5;
    const int  gwave  = blockIdx.x * WPB + wave;
    const int  nwave  = gridDim.x * WPB;
    const int  n      = lane & 15;          // N column this lane owns (B/C/D)
    const bool hiHalf = lane >= 16;
    const int  hb     = hiHalf ? 8 : 0;

    // ---- Build the byte->8xf16 LUT (one 16B row per thread). ----
    {
        union { _Float16 h[8]; float4 f4; } u;
        const int t = threadIdx.x;          // 0..255 == byte value
#pragma unroll
        for (int j = 0; j < 8; ++j)
            u.h[j] = ((t >> j) & 1) ? (_Float16)1.0f : (_Float16)0.0f;
        lut[t] = u.f4;
    }
    __syncthreads();

    // ---- Build B fragments once per wave; resident in VGPRs thereafter. ----
    // Wcat[k][n]: n<4 -> zero-padded W0 (k>=64 is 0), n in 4..7 -> W1, n>=8 -> 0.
    // Dense 32x16 f16 B layout per chunk (matches A): elements 0..7 <-> K=hb+0..7,
    // elements 8..15 <-> K=16+hb+0..7, for this lane's N = lane%16.
    v16h bfrag[4];
#pragma unroll
    for (int c = 0; c < 4; ++c) {
#pragma unroll
        for (int v = 0; v < 8; ++v) {
#pragma unroll
            for (int o = 0; o < 2; ++o) {
                const int k = 32 * c + hb + (v >= 4 ? 8 : 0) + 2 * v + o;
                float w = 0.0f;
                if (n < 4)      w = (k < 64) ? W0[n * 64 + k] : 0.0f;
                else if (n < 8) w = W1[(n - 4) * 128 + k];
                bfrag[c][2 * v + o] = (_Float16)w;
            }
        }
    }
    // Bias: every accumulator element of this lane lives in column n.
    const float biasv = (n < 4) ? b0[n] : ((n < 8) ? b1[n - 4] : 0.0f);

    const _Float16* lth = (const _Float16*)lut;

    // ------------------------- main tile loop (16 samples / wave-iter) ------
    const int ntiles = nB >> 4;
    for (int tile = gwave; tile < ntiles; tile += nwave) {
        const int m = n;                    // A row this lane supplies
        const int s = tile * 16 + m;        // sample id (dup across lane halves)
        const int2 ii = ((const int2*)idxs)[s];
        const uint64_t m0 = 1ull << ii.x;   // one-hot bitmask, slot 0 (k 0..63)
        const uint64_t m1 = 1ull << ii.y;   // one-hot bitmask, slot 1 (k 64..127)

        // Epilogue operands (2 half-samples per lane) loaded early.
        const int sm = tile * 16 + (lane >> 1);
        const int ty = types[sm];

        v8f acc = {biasv, biasv, biasv, biasv, biasv, biasv, biasv, biasv};
#pragma unroll
        for (int c = 0; c < 4; ++c) {
            // A 16-bit 16x32 layout: elements 0..7 <-> K = 32c+hb+0..7,
            // elements 8..15 <-> K = 32c+16+hb+0..7 (runs of 8 consecutive K).
            const uint64_t mm = (c < 2) ? m0 : m1;
            const int base   = (c & 1) * 32 + hb;     // K offset within 64-range
            const unsigned bA = (unsigned)(mm >> base)        & 0xffu;
            const unsigned bB = (unsigned)(mm >> (base + 16)) & 0xffu;
            const v8h lo = *(const v8h*)(lth + bA * 8);   // ds_load_b128
            const v8h hi = *(const v8h*)(lth + bB * 8);   // ds_load_b128
            const v16h a = __builtin_shufflevector(lo, hi,
                0, 1, 2, 3, 4, 5, 6, 7, 8, 9, 10, 11, 12, 13, 14, 15);
            acc = __builtin_amdgcn_wmma_f32_16x16x32_f16(
                false, a, false, bfrag[c], (short)0, acc, false, false);
        }

        // Stage 16x16 f32 D tile to LDS; per-sample type-select + tanh + store.
        float* st = stage[wave];
#pragma unroll
        for (int r = 0; r < 8; ++r)
            st[(r + (hiHalf ? 8 : 0)) * 16 + n] = acc[r];
        asm volatile("s_wait_dscnt 0" ::: "memory");  // wave-internal LDS fence

        // 32 lanes cover 16 samples x 2 outputs each.
        const int jo = (lane & 1) * 2;
        const float2 yv = *(const float2*)(st + (lane >> 1) * 16 + ty * 4 + jo);
        float2 yo;
        yo.x = fast_tanh(yv.x);
        yo.y = fast_tanh(yv.y);
        *(float2*)(out + sm * 4 + jo) = yo;
    }

    // ------------------------- tail (nB % 16 != 0): exact scalar gather ----
    const int tail = ntiles << 4;
    for (int s = tail + blockIdx.x * blockDim.x + threadIdx.x; s < nB;
         s += gridDim.x * blockDim.x) {
        const int t  = types[s];
        const int i0 = idxs[2 * s + 0];
        const int i1 = idxs[2 * s + 1];
#pragma unroll
        for (int j = 0; j < 4; ++j) {
            const float y = (t == 0)
                ? (W0[j * 64 + i0] + b0[j])
                : (W1[j * 128 + i0] + W1[j * 128 + 64 + i1] + b1[j]);
            out[s * 4 + j] = fast_tanh(y);
        }
    }
}

extern "C" void kernel_launch(void* const* d_in, const int* in_sizes, int n_in,
                              void* d_out, int out_size, void* d_ws, size_t ws_size,
                              hipStream_t stream) {
    (void)n_in; (void)d_ws; (void)ws_size; (void)out_size;
    const int*   idxs  = (const int*)  d_in[0];  // action_indecies [B,2]
    // d_in[1] (action_n_obj) is derivable from action_types; unused.
    const int*   types = (const int*)  d_in[2];  // action_types [B]
    const float* W0    = (const float*)d_in[3];  // [4,64]
    const float* b0    = (const float*)d_in[4];  // [4]
    const float* W1    = (const float*)d_in[5];  // [4,128]
    const float* b1    = (const float*)d_in[6];  // [4]
    float*       out   = (float*)d_out;

    const int B = in_sizes[2];                   // number of samples
    const int blocks = 2048;                     // 16384 waves; grid-stride tiles
    action_encoder_wmma<<<blocks, 256, 0, stream>>>(idxs, types, W0, b0, W1, b1,
                                                    out, B);
}
